// HetCAN_8890582302908
// MI455X (gfx1250) — matrix operand order repaired and checked by reading, not compiled, not run on hardware
//
#include <hip/hip_runtime.h>
#include <math.h>

// ---------------------------------------------------------------------------
// HetCAN forward for MI455X (gfx1250, wave32).
//  - Dense GEMMs via v_wmma_f32_16x16x32_f16:
//      * weights pre-packed once into per-lane f16 B-fragment layout
//        (2x global_load_b128 per fragment, resident in 192MB L2),
//      * A chunk converted to the per-lane fragment image in LDS and
//        broadcast to all 8 waves of the block (ds_load_b128); A is read
//        from HBM exactly once per GEMM,
//      * B-fragment loads issued BEFORE the block barrier so the barrier +
//        LDS read hide the global-load latency of the WMMA operands.
//  - Edge softmax/aggregation: bandwidth-bound scatter (~2.4GB total) with
//    contiguous 128B feat rows + f32 atomics resolving in L2.
//  - Edge-feature term precomputed to a [layer][etype][head] scalar table.
//  - DAL transformer stack collapses to a single scalar shift (LN over a
//    singleton channel axis emits only its bias -> q/k/v constant, attention
//    uniform, FFN constant). Computed faithfully by one thread (exact GELU).
// ---------------------------------------------------------------------------

#define NN   24000
#define EE   384000
#define NTY  3
#define INF_ 128
#define HID  32
#define EDIM 64
#define NEF  5
#define FFN_ 64
#define CLS  16

typedef __attribute__((ext_vector_type(16))) _Float16 v16h;
typedef __attribute__((ext_vector_type(8)))  float    v8f;

static inline int cdiv(int a, int b) { return (a + b - 1) / b; }

// ---------------------------------------------------------------------------
// Pack B[K,N] (f32, row-major) into per-lane WMMA fragment layout (f16):
// frag index = ((kc*tilesN + tn)*32 + lane), 16 halves (32B) contiguous.
// Lane holds column col=(tn*16 + lane%16); half e -> K = kc*32 + 16*(lane/16) + e.
// ---------------------------------------------------------------------------
__global__ void k_pack_b(const float* __restrict__ B, unsigned int* __restrict__ out,
                         int K, int Nn)
{
    const int tilesN = Nn >> 4;
    const int total  = (K >> 5) * tilesN * 32;
    const int idx = blockIdx.x * blockDim.x + threadIdx.x;
    if (idx >= total) return;
    const int lane = idx & 31;
    const int t    = idx >> 5;
    const int tn   = t % tilesN;
    const int kc   = t / tilesN;
    const int g    = lane >> 4;
    const int col  = (tn << 4) + (lane & 15);
    const float* src = B + (size_t)((kc << 5) + 16 * g) * Nn + col;
    unsigned int* dst = out + (size_t)idx * 8;
#pragma unroll
    for (int p = 0; p < 8; ++p) {
        union { unsigned int u; _Float16 h[2]; } cv;
        cv.h[0] = (_Float16)src[(size_t)(2 * p) * Nn];
        cv.h[1] = (_Float16)src[(size_t)(2 * p + 1) * Nn];
        dst[p] = cv.u;
    }
}

// ---------------------------------------------------------------------------
// WMMA GEMM: C[M,N] = act(A[M,K] @ B[K,N] + bias) (* mul)
// act: 0=none, 1=leaky_relu(0.01), 2=relu.  M%16==0, N%16==0 (N<=256), K%32==0.
// One block (8 waves) = one 16-row stripe; waves cover column tiles
// {wave, wave+8}; A fragment staged in LDS (double-buffered, one barrier per
// k-step); B pre-packed in fragment layout and fetched ahead of the barrier.
// ---------------------------------------------------------------------------
__global__ __launch_bounds__(256)
void k_wmma_gemm(const float* __restrict__ A, const _Float16* __restrict__ Bp,
                 const float* __restrict__ bias, const float* __restrict__ mul,
                 float* __restrict__ C, int M, int Nn, int K, int act)
{
    __shared__ __align__(32) unsigned int ldsA[2][32][8];   // per-lane A fragment image

    const int row0   = blockIdx.x << 4;
    const int tid    = threadIdx.x;
    const int wave   = tid >> 5;
    const int lane   = tid & 31;
    const int g      = lane >> 4;
    const int ml     = lane & 15;
    const int tilesN = Nn >> 4;

    // column-tile assignment (wave-uniform; clamped so loads never diverge)
    const int  tn0 = wave, tn1 = wave + 8;
    const bool do0 = tn0 < tilesN, do1 = tn1 < tilesN;
    const int  ctn0 = do0 ? tn0 : 0;
    const int  ctn1 = do1 ? tn1 : 0;

    // staging role: thread t writes halves (2*se, 2*se+1) of lane image sl
    const int sl  = tid >> 3;            // 0..31
    const int se  = tid & 7;             // uint slot 0..7
    const int sg  = sl >> 4, sml = sl & 15;
    const int e0  = 2 * se;              // half index within fragment
    const int kof = 16 * sg + e0;

    v8f acc0 = {}, acc1 = {};
    const int nk = K >> 5;
    for (int kc = 0; kc < nk; ++kc) {
        // ---- stage A chunk (16 rows x 32 cols, f32 -> f16 fragment image) ----
        const float* ap = A + (size_t)(row0 + sml) * K + (kc << 5) + kof;
        union { unsigned int u; _Float16 h[2]; } cv;
        cv.h[0] = (_Float16)ap[0];
        cv.h[1] = (_Float16)ap[1];
        if (kc + 1 < nk) __builtin_prefetch(ap + 32, 0, 1);   // global_prefetch_b8
        ldsA[kc & 1][sl][se] = cv.u;

        // ---- issue B fragment loads BEFORE the barrier (latency hiding) ----
        const v16h bf0 = *reinterpret_cast<const v16h*>(
            Bp + (((size_t)kc * tilesN + ctn0) * 32 + lane) * 16);
        const v16h bf1 = *reinterpret_cast<const v16h*>(
            Bp + (((size_t)kc * tilesN + ctn1) * 32 + lane) * 16);

        __syncthreads();

        // ---- A fragment: 32B broadcast read from LDS (2x ds_load_b128) ----
        const v16h afrag = *reinterpret_cast<const v16h*>(&ldsA[kc & 1][lane][0]);

        // ---- WMMA (guards are wave-uniform -> EXEC all-1s when executed) ----
        if (do0)
            acc0 = __builtin_amdgcn_wmma_f32_16x16x32_f16(
                       false, afrag, false, bf0, (short)0, acc0, false, false);
        if (do1)
            acc1 = __builtin_amdgcn_wmma_f32_16x16x32_f16(
                       false, afrag, false, bf1, (short)0, acc1, false, false);
    }

    // ---- epilogue ----
#pragma unroll
    for (int tt = 0; tt < 2; ++tt) {
        const int tn = wave + tt * 8;
        if (tn >= tilesN) continue;
        const v8f acc = tt ? acc1 : acc0;
        const int col = (tn << 4) + ml;
        const float bv = bias ? bias[col] : 0.0f;
#pragma unroll
        for (int r = 0; r < 8; ++r) {
            const int row = row0 + r + 8 * g;     // C/D layout: VGPR r -> M = r + 8g
            float v = acc[r] + bv;
            if (act == 1)      v = (v >= 0.0f) ? v : 0.01f * v;
            else if (act == 2) v = fmaxf(v, 0.0f);
            if (mul) v *= mul[(size_t)row * Nn + col];
            C[(size_t)row * Nn + col] = v;
        }
    }
}

// ---------------------------------------------------------------------------
// Small precomputed tables: eterm[l][etype][head] and the DAL scalar shift.
// ---------------------------------------------------------------------------
__global__ void k_tables(const float* __restrict__ ee0, const float* __restrict__ We0, const float* __restrict__ ae0,
                         const float* __restrict__ ee1, const float* __restrict__ We1, const float* __restrict__ ae1,
                         const float* __restrict__ ee2, const float* __restrict__ We2, const float* __restrict__ ae2,
                         const float* __restrict__ ln1b, const float* __restrict__ Wv,  const float* __restrict__ bvv,
                         const float* __restrict__ Wo,   const float* __restrict__ bo,
                         const float* __restrict__ ln2b, const float* __restrict__ W1,  const float* __restrict__ b1,
                         const float* __restrict__ W2,   const float* __restrict__ b2,
                         float* __restrict__ eterm, float* __restrict__ da_shift)
{
    const int tid = threadIdx.x;
    if (tid < 3 * NEF * 8) {                    // 120 (layer, etype, head) slots
        const int l  = tid / (NEF * 8);
        const int r  = tid - l * (NEF * 8);
        const int t  = r >> 3;
        const int hh = r & 7;
        const float* ee; const float* We; const float* ae; int Hl;
        if (l == 0)      { ee = ee0; We = We0; ae = ae0; Hl = 8; }
        else if (l == 1) { ee = ee1; We = We1; ae = ae1; Hl = 8; }
        else             { ee = ee2; We = We2; ae = ae2; Hl = 1; }
        float s = 0.0f;
        if (hh < Hl) {
            for (int d = 0; d < EDIM; ++d) {
                float proj = 0.0f;
                for (int c = 0; c < EDIM; ++c)
                    proj += ee[t * EDIM + c] * We[c * (Hl * EDIM) + hh * EDIM + d];
                s += proj * ae[hh * EDIM + d];
            }
        }
        eterm[(l * NEF + t) * 8 + hh] = s;
    } else if (tid == 120) {
        // DAL stack: LN over singleton channel -> output == bias; attention
        // uniform; whole stack is x += shift (shift independent of x).
        float s = 0.0f;
        for (int l = 0; l < 2; ++l) {
            const float y1 = ln1b[l];
            const float v  = y1 * Wv[l] + bvv[l];        // uniform att * v == v
            s += v * Wo[l] + bo[l];
            const float y2 = ln2b[l];
            float add2 = b2[l];
            for (int f = 0; f < FFN_; ++f) {
                const float t = y2 * W1[l * FFN_ + f] + b1[l * FFN_ + f];
                const float ge = 0.5f * t * (1.0f + erff(t * 0.70710678118654752f));
                add2 += ge * W2[l * FFN_ + f];
            }
            s += add2;
        }
        *da_shift = s;
    }
}

// ntype feature per node: ntf[n][j] = (ntype_emb[node_type[n]] @ W_ntype)[j]
__global__ void k_ntf(const float* __restrict__ ntype_emb, const float* __restrict__ W_ntype,
                      const int* __restrict__ node_type, float* __restrict__ ntf)
{
    const int idx = blockIdx.x * blockDim.x + threadIdx.x;
    if (idx >= NN * HID) return;
    const int n = idx >> 5, j = idx & 31;
    const int t = node_type[n];
    float s = 0.0f;
    for (int c = 0; c < HID; ++c) s += ntype_emb[t * HID + c] * W_ntype[c * HID + j];
    ntf[idx] = s;
}

__global__ void k_fill(float* __restrict__ p, int n, float v)
{
    const int idx = blockIdx.x * blockDim.x + threadIdx.x;
    if (idx < n) p[idx] = v;
}

// el/er per (node, head) + init of segment-max / denom accumulators.
__global__ void k_el_er(const float* __restrict__ feat, const float* __restrict__ al,
                        const float* __restrict__ ar, float* __restrict__ el,
                        float* __restrict__ er, float* __restrict__ mmax,
                        float* __restrict__ denom, int H)
{
    const int idx = blockIdx.x * blockDim.x + threadIdx.x;
    if (idx >= NN * H) return;
    const int n = idx / H, hh = idx - n * H;
    const float* f = feat + (size_t)n * H * HID + hh * HID;
    float sl = 0.0f, sr = 0.0f;
#pragma unroll
    for (int d = 0; d < HID; ++d) { sl += f[d] * al[hh * HID + d]; sr += f[d] * ar[hh * HID + d]; }
    el[idx] = sl; er[idx] = sr;
    mmax[idx]  = -INFINITY;
    denom[idx] = 0.0f;
}

// sign-aware float atomic max (valid for non-NaN, init = -inf)
__device__ inline void atomicMaxFloat(float* addr, float v)
{
    if (v >= 0.0f) atomicMax((int*)addr, __float_as_int(v));
    else           atomicMin((unsigned int*)addr, __float_as_uint(v));
}

// Pass 1: logits + segment max over dst.
__global__ void k_edge_max(const float* __restrict__ el, const float* __restrict__ er,
                           const float* __restrict__ eterm_l, const int* __restrict__ e_feat,
                           const int* __restrict__ src, const int* __restrict__ dst,
                           float* __restrict__ lg, float* __restrict__ mmax, int H)
{
    const int e = blockIdx.x * blockDim.x + threadIdx.x;
    if (e >= EE) return;
    const int s = src[e], d = dst[e], ef = e_feat[e];
    for (int hh = 0; hh < H; ++hh) {
        float z = el[s * H + hh] + er[d * H + hh] + eterm_l[ef * 8 + hh];
        z = (z >= 0.0f) ? z : 0.2f * z;            // leaky_relu(0.2)
        lg[(size_t)e * H + hh] = z;
        atomicMaxFloat(&mmax[d * H + hh], z);
    }
}

// Pass 2: exp(lg - max) + segment sum.
__global__ void k_edge_exp(const float* __restrict__ lg, const float* __restrict__ mmax,
                           const int* __restrict__ dst, float* __restrict__ a_cur,
                           float* __restrict__ denom, int H)
{
    const int idx = blockIdx.x * blockDim.x + threadIdx.x;
    if (idx >= EE * H) return;
    const int e = idx / H, hh = idx - e * H;
    const float w = expf(lg[idx] - mmax[dst[e] * H + hh]);
    a_cur[idx] = w;
    atomicAdd(&denom[dst[e] * H + hh], w);
}

// Pass 3: normalize, residual-attention mix, weighted scatter-add of feat[src].
__global__ void k_edge_aggr(const float* __restrict__ a_cur, const float* __restrict__ denom,
                            float* __restrict__ a_save, const float* __restrict__ a_res,
                            const int* __restrict__ src, const int* __restrict__ dst,
                            const float* __restrict__ feat, float* __restrict__ rst, int H)
{
    const int idx = blockIdx.x * blockDim.x + threadIdx.x;
    if (idx >= EE * H) return;
    const int e = idx / H, hh = idx - e * H;
    float a = a_cur[idx] / denom[dst[e] * H + hh];
    if (a_res)  a = a * 0.95f + a_res[idx] * 0.05f;   // (1-ALPHA), ALPHA
    if (a_save) a_save[idx] = a;
    const float* fs = feat + (size_t)src[e] * H * HID + hh * HID;   // 128B contiguous
    float*       rd = rst  + (size_t)dst[e] * H * HID + hh * HID;
#pragma unroll
    for (int d = 0; d < HID; ++d) atomicAdd(rd + d, a * fs[d]);
}

// Node epilogue: out = act(rst + bias (+ res))
__global__ void k_node_ep(const float* __restrict__ rst, const float* __restrict__ res,
                          const float* __restrict__ bias, float* __restrict__ out,
                          int H, int elu)
{
    const int idx = blockIdx.x * blockDim.x + threadIdx.x;
    if (idx >= NN * H * HID) return;
    const int r = idx % (H * HID);
    float v = rst[idx] + bias[r] + (res ? res[idx] : 0.0f);
    if (elu) v = (v > 0.0f) ? v : expm1f(v);
    out[idx] = v;
}

// hcat = [ l2n(h_feat) , l2n(h_feat*ntf + da_shift) ]
__global__ void k_build_hcat(const float* __restrict__ h_feat, const float* __restrict__ ntf,
                             const float* __restrict__ da_shift, float* __restrict__ hcat)
{
    const int n = blockIdx.x * blockDim.x + threadIdx.x;
    if (n >= NN) return;
    const float shift = *da_shift;
    const float* hf = h_feat + (size_t)n * HID;
    const float* nt = ntf    + (size_t)n * HID;
    float xv[HID];
    float s1 = 0.0f, s2 = 0.0f;
#pragma unroll
    for (int i = 0; i < HID; ++i) {
        const float a = hf[i];
        const float x = a * nt[i] + shift;
        xv[i] = x; s1 += a * a; s2 += x * x;
    }
    const float n1 = fmaxf(sqrtf(s1), 1e-12f);
    const float n2 = fmaxf(sqrtf(s2), 1e-12f);
#pragma unroll
    for (int i = 0; i < HID; ++i) {
        hcat[(size_t)n * 64 + i]      = hf[i] / n1;
        hcat[(size_t)n * 64 + 32 + i] = xv[i] / n2;
    }
}

// Row L2-normalize logits in place.
__global__ void k_row_norm(float* __restrict__ o)
{
    const int n = blockIdx.x * blockDim.x + threadIdx.x;
    if (n >= NN) return;
    float* r = o + (size_t)n * CLS;
    float s = 0.0f;
#pragma unroll
    for (int j = 0; j < CLS; ++j) s += r[j] * r[j];
    const float inv = 1.0f / fmaxf(sqrtf(s), 1e-12f);
#pragma unroll
    for (int j = 0; j < CLS; ++j) r[j] *= inv;
}

// ---------------------------------------------------------------------------
extern "C" void kernel_launch(void* const* d_in, const int* in_sizes, int n_in,
                              void* d_out, int out_size, void* d_ws, size_t ws_size,
                              hipStream_t stream)
{
    (void)in_sizes; (void)n_in; (void)out_size; (void)ws_size;
    const float* feats     = (const float*)d_in[0];
    const int*   e_feat    = (const int*)d_in[1];
    const int*   node_type = (const int*)d_in[2];
    const int*   src       = (const int*)d_in[3];
    const int*   dst       = (const int*)d_in[4];
    const float* fc_W      = (const float*)d_in[5];
    const float* fc_b      = (const float*)d_in[6];
    const float* ntype_emb = (const float*)d_in[7];
    const float* W_ntype   = (const float*)d_in[8];
    // ta0: 9..15  | ta1: 16..23 | ta2: 24..31
    const float* ta0_ee = (const float*)d_in[9],  *ta0_W = (const float*)d_in[10],
               * ta0_We = (const float*)d_in[11], *ta0_al = (const float*)d_in[12],
               * ta0_ar = (const float*)d_in[13], *ta0_ae = (const float*)d_in[14],
               * ta0_bias = (const float*)d_in[15];
    const float* ta1_ee = (const float*)d_in[16], *ta1_W = (const float*)d_in[17],
               * ta1_We = (const float*)d_in[18], *ta1_al = (const float*)d_in[19],
               * ta1_ar = (const float*)d_in[20], *ta1_ae = (const float*)d_in[21],
               * ta1_bias = (const float*)d_in[22], *ta1_res_W = (const float*)d_in[23];
    const float* ta2_ee = (const float*)d_in[24], *ta2_W = (const float*)d_in[25],
               * ta2_We = (const float*)d_in[26], *ta2_al = (const float*)d_in[27],
               * ta2_ar = (const float*)d_in[28], *ta2_ae = (const float*)d_in[29],
               * ta2_bias = (const float*)d_in[30], *ta2_res_W = (const float*)d_in[31];
    const float* da_ln1_b = (const float*)d_in[33];
    const float* da_Wv    = (const float*)d_in[38];
    const float* da_bv    = (const float*)d_in[39];
    const float* da_Wo    = (const float*)d_in[40];
    const float* da_bo    = (const float*)d_in[41];
    const float* da_ln2_b = (const float*)d_in[43];
    const float* da_W1    = (const float*)d_in[44];
    const float* da_b1    = (const float*)d_in[45];
    const float* da_W2    = (const float*)d_in[46];
    const float* da_b2    = (const float*)d_in[47];
    const float* final_W  = (const float*)d_in[48];
    const float* logits_W = (const float*)d_in[49];

    // workspace bump allocator (32B-aligned chunks)
    float* ws = (float*)d_ws;
    size_t off = 0;
    auto alloc = [&](size_t n) { float* p = ws + off; off += (n + 7) & ~(size_t)7; return p; };
    float* eterm    = alloc(3 * NEF * 8);
    float* da_shift = alloc(8);
    float* ntf      = alloc((size_t)NN * HID);
    float* h_a      = alloc((size_t)NN * 256);
    float* h_b      = alloc((size_t)NN * 256);
    float* featb    = alloc((size_t)NN * 256);
    float* resb     = alloc((size_t)NN * 256);
    float* rst      = alloc((size_t)NN * 256);
    float* el       = alloc((size_t)NN * 8);
    float* er       = alloc((size_t)NN * 8);
    float* mmax     = alloc((size_t)NN * 8);
    float* denom    = alloc((size_t)NN * 8);
    float* lg       = alloc((size_t)EE * 8);
    float* a_cur    = alloc((size_t)EE * 8);
    float* a0       = alloc((size_t)EE * 8);
    float* h_feat   = alloc((size_t)NN * HID);
    float* hcat     = alloc((size_t)NN * 64);
    float* t1       = alloc((size_t)NN * HID);
    float* bpack    = alloc(256 * 256 / 2);     // f16 packed weights (max 128KB)

    auto gemm = [&](const float* A, const float* B, const float* bias, const float* mul,
                    float* C, int M, int Nn, int K, int act) {
        const int frags = (K / 32) * (Nn / 16) * 32;
        k_pack_b<<<cdiv(frags, 256), 256, 0, stream>>>(B, (unsigned int*)bpack, K, Nn);
        k_wmma_gemm<<<M / 16, 256, 0, stream>>>(A, (const _Float16*)bpack,
                                                bias, mul, C, M, Nn, K, act);
    };

    // --- small tables + per-node ntype feature --------------------------------
    k_tables<<<1, 128, 0, stream>>>(ta0_ee, ta0_We, ta0_ae, ta1_ee, ta1_We, ta1_ae,
                                    ta2_ee, ta2_We, ta2_ae,
                                    da_ln1_b, da_Wv, da_bv, da_Wo, da_bo,
                                    da_ln2_b, da_W1, da_b1, da_W2, da_b2,
                                    eterm, da_shift);
    k_ntf<<<cdiv(NN * HID, 256), 256, 0, stream>>>(ntype_emb, W_ntype, node_type, ntf);

    // --- input projection (per type), fused lrelu(0.01) * ntype_feat ----------
    for (int t = 0; t < NTY; ++t) {
        const int rows = NN / NTY;
        gemm(feats + (size_t)t * rows * INF_, fc_W + (size_t)t * INF_ * HID,
             fc_b + t * HID, ntf + (size_t)t * rows * HID,
             h_a + (size_t)t * rows * HID, rows, HID, INF_, /*lrelu*/1);
    }

    // --- three TA layers ------------------------------------------------------
    auto ta_layer = [&](int l, const float* h_in, int di, int H,
                        const float* Wl, const float* res_Wl,
                        const float* al, const float* ar, const float* biasl,
                        float* h_out, float* a_save, const float* a_res, int elu) {
        gemm(h_in, Wl, nullptr, nullptr, featb, NN, H * HID, di, 0);
        if (res_Wl) gemm(h_in, res_Wl, nullptr, nullptr, resb, NN, H * HID, di, 0);
        k_el_er<<<cdiv(NN * H, 256), 256, 0, stream>>>(featb, al, ar, el, er, mmax, denom, H);
        k_fill<<<cdiv(NN * H * HID, 256), 256, 0, stream>>>(rst, NN * H * HID, 0.0f);
        k_edge_max<<<cdiv(EE, 256), 256, 0, stream>>>(el, er, eterm + l * NEF * 8,
                                                      e_feat, src, dst, lg, mmax, H);
        k_edge_exp<<<cdiv(EE * H, 256), 256, 0, stream>>>(lg, mmax, dst, a_cur, denom, H);
        k_edge_aggr<<<cdiv(EE * H, 256), 256, 0, stream>>>(a_cur, denom, a_save, a_res,
                                                           src, dst, featb, rst, H);
        k_node_ep<<<cdiv(NN * H * HID, 256), 256, 0, stream>>>(
            rst, res_Wl ? resb : nullptr, biasl, h_out, H, elu);
    };

    ta_layer(0, h_a, HID, 8, ta0_W, nullptr,    ta0_al, ta0_ar, ta0_bias, h_b,    a0,      nullptr, 1);
    ta_layer(1, h_b, 256, 8, ta1_W, ta1_res_W,  ta1_al, ta1_ar, ta1_bias, h_a,    nullptr, a0,      1);
    ta_layer(2, h_a, 256, 1, ta2_W, ta2_res_W,  ta2_al, ta2_ar, ta2_bias, h_feat, nullptr, nullptr, 0);

    // --- DAL-collapsed head ---------------------------------------------------
    k_build_hcat<<<cdiv(NN, 256), 256, 0, stream>>>(h_feat, ntf, da_shift, hcat);
    gemm(hcat, final_W, nullptr, nullptr, t1, NN, HID, 64, /*relu*/2);
    gemm(t1, logits_W, nullptr, nullptr, (float*)d_out, NN, CLS, HID, 0);
    k_row_norm<<<cdiv(NN, 256), 256, 0, stream>>>((float*)d_out);
}
